// GINFeatures_2018634629569
// MI455X (gfx1250) — matrix operand order repaired and checked by reading, not compiled, not run on hardware
//
#include <hip/hip_runtime.h>
#include <hip/hip_bf16.h>
#include <math.h>

// ---------------------------------------------------------------------------
// GIN forward on MI455X (gfx1250, wave32, WMMA).
// GEMMs: v_wmma_f32_16x16x32_bf16, 2x2 tiles/wave,
//        async global->LDS staging (ASYNCcnt), b128 LDS fragment loads.
// ---------------------------------------------------------------------------

typedef __bf16 bf16_t;
typedef __attribute__((address_space(3))) __bf16 lds_bf16_t;
typedef __attribute__((ext_vector_type(16))) __bf16 v16bf;
typedef __attribute__((ext_vector_type(8)))  __bf16 v8bf;   // one b128
typedef __attribute__((ext_vector_type(8)))  float  v8f;

#define NN      50000
#define NE      150000
#define FIN     128
#define DD      512
#define NG      256
#define BN_EPS  1e-5f

// async copy of 16B global -> LDS, no VGPR round trip (ASYNCcnt-tracked)
static __device__ __forceinline__ void async_ld_b128(bf16_t* lds_dst,
                                                     const bf16_t* gsrc) {
  unsigned l = (unsigned)(unsigned long long)(lds_bf16_t*)lds_dst;
  asm volatile("global_load_async_to_lds_b128 %0, %1, off"
               :: "v"(l), "v"((unsigned long long)gsrc) : "memory");
}
static __device__ __forceinline__ void wait_async0() {
  asm volatile("s_wait_asynccnt 0x0" ::: "memory");
}

// ---------------------------------------------------------------- elementwise
__global__ __launch_bounds__(256) void copy_f32_k(const float* __restrict__ in,
                                                  float* __restrict__ out, size_t n) {
  size_t i = (size_t)blockIdx.x * 256 + threadIdx.x;
  if (i < n) out[i] = in[i];
}

__global__ __launch_bounds__(256) void f32_to_bf16_k(const float* __restrict__ in,
                                                     bf16_t* __restrict__ out, size_t n) {
  size_t i = (size_t)blockIdx.x * 256 + threadIdx.x;
  if (i < n) out[i] = (bf16_t)in[i];
}

// weights: f32 [K][N] -> bf16 transposed [N][K] (done once per layer, tiny)
__global__ __launch_bounds__(256) void f32_to_bf16_T_k(const float* __restrict__ in,
                                                       bf16_t* __restrict__ out,
                                                       int K, int N) {
  int i = blockIdx.x * 256 + threadIdx.x;
  if (i >= K * N) return;
  int n = i / K, k = i % K;       // output-coalesced
  out[i] = (bf16_t)in[(size_t)k * N + n];
}

// h_sum[dst] += x[src] over all edges (edge x feature parallel, L2 atomics)
__global__ __launch_bounds__(256) void edge_agg_k(const float* __restrict__ x,
                                                  const long long* __restrict__ ei,
                                                  float* __restrict__ hs, int F) {
  size_t i = (size_t)blockIdx.x * 256 + threadIdx.x;
  size_t total = (size_t)NE * F;
  if (i >= total) return;
  int e = (int)(i / F);
  int f = (int)(i % F);
  long long s = ei[e];              // edge_index row 0 = src
  long long d = ei[(size_t)NE + e]; // edge_index row 1 = dst
  atomicAdd(&hs[(size_t)d * F + f], x[(size_t)s * F + f]);
}

// ---------------------------------------------------------------- WMMA GEMM
// C[M,N] = epi( A[M,K](bf16 rm) @ BT[N,K](bf16 rm, pre-transposed) + bias[N] )
// epi: 0=none, 1=relu, 2=tanh; out_bf16: write Ch (bf16) instead of Cf (f32).
// Block: 256 thr = 8 waves; tile 128(M) x 64(N); wave = 32x32 patch = 2x2 WMMA.
// LDS row stride 40 bf16 (80B = 20 banks): b128 fragment loads tile all banks.
#define BM 128
#define BN 64
#define BK 32
#define LDSTR 40

__global__ __launch_bounds__(256)
void gemm_wmma_bf16_k(const bf16_t* __restrict__ A, const bf16_t* __restrict__ BT,
                      const float* __restrict__ bias,
                      float* __restrict__ Cf, bf16_t* __restrict__ Ch,
                      int M, int K, int N, int epi, int out_bf16) {
  __shared__ bf16_t sA [BM * LDSTR];   // A tile  [m][k]
  __shared__ bf16_t sBT[BN * LDSTR];   // B tile  [n][k] (already transposed)

  const int tid  = threadIdx.x;
  const int lane = tid & 31;
  const int wave = tid >> 5;     // 0..7
  const int wm   = wave >> 1;    // 0..3 -> M offset wm*32
  const int wn   = wave & 1;     // 0..1 -> N offset wn*32
  const int m0   = blockIdx.x * BM;
  const int n0   = blockIdx.y * BN;

  // fragment addressing per ISA 16-bit layouts (wave32):
  // A: lane -> row m=lane&15, K runs [kg,kg+8) and [kg+16,kg+24), kg=(lane>>4)*8
  // B(^T): lane -> col n=lane&15, K runs [kb0,kb0+8) and [kb0+16,kb0+24), kb0=(lane>>4)*8
  const int mA  = lane & 15;
  const int kg  = (lane >> 4) * 8;
  const int nB  = lane & 15;
  const int kb0 = (lane >> 4) * 8;

  // per-thread staging assignments (pure contiguous b128 copies)
  const int rA    = tid >> 1;           // 0..127 A row
  const int hA    = (tid & 1) * 16;     // 0 / 16 A col half
  const int rB    = tid >> 2;           // 0..63  BT row
  const int hB    = (tid & 3) * 8;      // 0/8/16/24 BT col chunk

  v8f acc[2][2] = {};

  for (int k0 = 0; k0 < K; k0 += BK) {
    // ---- stage A: 128x32, two async b128 per thread
    {
      int gm = m0 + rA;
      bf16_t* dst = &sA[rA * LDSTR + hA];
      if (gm < M) {
        const bf16_t* src = &A[(size_t)gm * K + k0 + hA];
        async_ld_b128(dst,     src);
        async_ld_b128(dst + 8, src + 8);
      } else {
        v8bf z = {};
        *(v8bf*)dst = z; *(v8bf*)(dst + 8) = z;
      }
    }
    // ---- stage BT: 64x32, one async b128 per thread (no transpose work:
    //      weights were pre-transposed once per layer)
    async_ld_b128(&sBT[rB * LDSTR + hB],
                  &BT[(size_t)(n0 + rB) * K + k0 + hB]);
    wait_async0();
    __syncthreads();

    // ---- fragments: 2x ds_load_b128 each
    v16bf af[2], bf2[2];
#pragma unroll
    for (int mi = 0; mi < 2; ++mi) {
      const bf16_t* p = &sA[(wm * 32 + mi * 16 + mA) * LDSTR + kg];
      v8bf lo = *(const v8bf*)p;
      v8bf hi = *(const v8bf*)(p + 16);
      af[mi] = __builtin_shufflevector(lo, hi, 0,1,2,3,4,5,6,7,8,9,10,11,12,13,14,15);
    }
#pragma unroll
    for (int ni = 0; ni < 2; ++ni) {
      const bf16_t* p = &sBT[(wn * 32 + ni * 16 + nB) * LDSTR + kb0];
      v8bf lo = *(const v8bf*)p;
      v8bf hi = *(const v8bf*)(p + 16);
      bf2[ni] = __builtin_shufflevector(lo, hi, 0,1,2,3,4,5,6,7,8,9,10,11,12,13,14,15);
    }
    // ---- 4 WMMAs per K step per wave
#pragma unroll
    for (int mi = 0; mi < 2; ++mi)
#pragma unroll
      for (int ni = 0; ni < 2; ++ni)
        acc[mi][ni] = __builtin_amdgcn_wmma_f32_16x16x32_bf16(
            false, af[mi], false, bf2[ni], (short)0, acc[mi][ni], false, false);
    __syncthreads();
  }

  // ---- epilogue: C layout VGPR r -> M=r+8*(lane>>4), N=lane&15
#pragma unroll
  for (int ni = 0; ni < 2; ++ni) {
    const int col = n0 + wn * 32 + ni * 16 + (lane & 15);
    const float bv = bias[col];
#pragma unroll
    for (int mi = 0; mi < 2; ++mi) {
      const int rbase = m0 + wm * 32 + mi * 16 + 8 * (lane >> 4);
#pragma unroll
      for (int r = 0; r < 8; ++r) {
        int row = rbase + r;
        if (row < M) {
          float v = acc[mi][ni][r] + bv;
          if (epi == 1)      v = fmaxf(v, 0.0f);
          else if (epi == 2) v = tanhf(v);
          if (out_bf16) Ch[(size_t)row * N + col] = (bf16_t)v;
          else          Cf[(size_t)row * N + col] = v;
        }
      }
    }
  }
}

// ---------------------------------------------------------------- batchnorm
__global__ __launch_bounds__(256) void bn_stats_k(const float* __restrict__ t,
                                                  float* __restrict__ sums,
                                                  float* __restrict__ sq, int M) {
  const int c0 = threadIdx.x;
  const int c1 = threadIdx.x + 256;
  const int r0 = blockIdx.x * 512;
  const int r1 = min(r0 + 512, M);
  float s0 = 0.f, s1 = 0.f, q0 = 0.f, q1 = 0.f;
  for (int r = r0; r < r1; ++r) {
    float a = fmaxf(t[(size_t)r * DD + c0], 0.f);
    float b = fmaxf(t[(size_t)r * DD + c1], 0.f);
    s0 += a; q0 += a * a; s1 += b; q1 += b * b;
  }
  atomicAdd(&sums[c0], s0); atomicAdd(&sq[c0], q0);
  atomicAdd(&sums[c1], s1); atomicAdd(&sq[c1], q1);
}

__global__ __launch_bounds__(256) void bn_norm_k(const float* __restrict__ t,
                                                 const float* __restrict__ sums,
                                                 const float* __restrict__ sq,
                                                 const float* __restrict__ gamma,
                                                 const float* __restrict__ beta,
                                                 float* __restrict__ out, int M) {
  size_t i = (size_t)blockIdx.x * 256 + threadIdx.x;
  size_t total = (size_t)M * DD;
  if (i >= total) return;
  int c = (int)(i % DD);
  float invM = 1.0f / (float)M;
  float mean = sums[c] * invM;
  float var  = sq[c] * invM - mean * mean;   // biased var == E[h^2]-mean^2
  float h    = fmaxf(t[i], 0.f);
  out[i] = gamma[c] * (h - mean) * rsqrtf(var + BN_EPS) + beta[c];
}

// ---------------------------------------------------------------- pooling
__global__ __launch_bounds__(256) void pool_accum_k(const float* __restrict__ h,
                                                    const long long* __restrict__ batch,
                                                    float* __restrict__ pool,
                                                    float* __restrict__ cnt) {
  size_t i = (size_t)blockIdx.x * 256 + threadIdx.x;
  size_t total = (size_t)NN * DD;
  if (i >= total) return;
  int r = (int)(i / DD);
  int c = (int)(i % DD);
  long long g = batch[r];
  atomicAdd(&pool[(size_t)g * DD + c], h[i]);
  if (c == 0) atomicAdd(&cnt[g], 1.0f);
}

__global__ __launch_bounds__(256) void pool_final_k(const float* __restrict__ pool,
                                                    const float* __restrict__ cnt,
                                                    bf16_t* __restrict__ out16) {
  size_t i = (size_t)blockIdx.x * 256 + threadIdx.x;
  size_t total = (size_t)NG * DD;
  if (i >= total) return;
  int g = (int)(i / DD);
  out16[i] = (bf16_t)(pool[i] / fmaxf(cnt[g], 1.0f));
}

// ---------------------------------------------------------------- launcher
static inline unsigned gblocks(size_t n) { return (unsigned)((n + 255) / 256); }

extern "C" void kernel_launch(void* const* d_in, const int* in_sizes, int n_in,
                              void* d_out, int out_size, void* d_ws, size_t ws_size,
                              hipStream_t stream) {
  const float*     x      = (const float*)d_in[0];
  const long long* ei     = (const long long*)d_in[1];   // int64 per reference
  const long long* batch  = (const long long*)d_in[2];
  const float*     W1a    = (const float*)d_in[3];
  const float*     b1a    = (const float*)d_in[4];
  const float*     W1b    = (const float*)d_in[5];
  const float*     b1b    = (const float*)d_in[6];
  const float*     Wa     = (const float*)d_in[7];   // (4, D, D)
  const float*     ba     = (const float*)d_in[8];   // (4, D)
  const float*     Wb     = (const float*)d_in[9];
  const float*     bb     = (const float*)d_in[10];
  const float*     gammas = (const float*)d_in[11];  // (5, D)
  const float*     betas  = (const float*)d_in[12];
  const float*     fcW    = (const float*)d_in[13];
  const float*     fcb    = (const float*)d_in[14];
  float*           out    = (float*)d_out;           // (256, 512) f32

  // ---- workspace carve (~310 MB) ----
  char* ws = (char*)d_ws;
  size_t off = 0;
  auto take = [&](size_t bytes) -> char* {
    char* p = ws + off;
    off = (off + bytes + 255) & ~(size_t)255;
    return p;
  };
  float*  bufS  = (float*)take((size_t)NN * DD * sizeof(float));   // h+agg / mlp2 out
  float*  bufH  = (float*)take((size_t)NN * DD * sizeof(float));   // block output
  bf16_t* bufA  = (bf16_t*)take((size_t)NN * DD * sizeof(bf16_t)); // bf16 act in
  bf16_t* bufB  = (bf16_t*)take((size_t)NN * DD * sizeof(bf16_t)); // bf16 act mid
  bf16_t* bufW  = (bf16_t*)take((size_t)DD * DD * sizeof(bf16_t)); // bf16 W^T
  float*  sums  = (float*)take(DD * sizeof(float));
  float*  sq    = (float*)take(DD * sizeof(float));
  float*  pool  = (float*)take((size_t)NG * DD * sizeof(float));
  float*  cnt   = (float*)take(NG * sizeof(float));
  bf16_t* p16   = (bf16_t*)take((size_t)NG * DD * sizeof(bf16_t));
  (void)ws_size; (void)in_sizes; (void)n_in; (void)out_size;

  const dim3 ggrid((NN + BM - 1) / BM, DD / BN);

  for (int layer = 0; layer < 5; ++layer) {
    const float* hin  = (layer == 0) ? x : bufH;
    const int    F    = (layer == 0) ? FIN : DD;
    const float* Wa_p = (layer == 0) ? W1a : Wa + (size_t)(layer - 1) * DD * DD;
    const float* ba_p = (layer == 0) ? b1a : ba + (size_t)(layer - 1) * DD;
    const float* Wb_p = (layer == 0) ? W1b : Wb + (size_t)(layer - 1) * DD * DD;
    const float* bb_p = (layer == 0) ? b1b : bb + (size_t)(layer - 1) * DD;
    const float* g_p  = gammas + (size_t)layer * DD;
    const float* be_p = betas  + (size_t)layer * DD;

    const size_t nh = (size_t)NN * F;
    // h_sum = h + sum_{j->i} h_j
    copy_f32_k<<<gblocks(nh), 256, 0, stream>>>(hin, bufS, nh);
    edge_agg_k<<<gblocks((size_t)NE * F), 256, 0, stream>>>(hin, ei, bufS, F);
    // MLP1: relu(h_sum @ Wa + ba) -> bf16 directly (skips f32 round trip)
    f32_to_bf16_k<<<gblocks(nh), 256, 0, stream>>>(bufS, bufA, nh);
    f32_to_bf16_T_k<<<gblocks((size_t)F * DD), 256, 0, stream>>>(Wa_p, bufW, F, DD);
    gemm_wmma_bf16_k<<<ggrid, 256, 0, stream>>>(bufA, bufW, ba_p,
                                                nullptr, bufB, NN, F, DD, /*relu*/1, 1);
    // MLP2: (.) @ Wb + bb -> f32 (feeds BN stats)
    f32_to_bf16_T_k<<<gblocks((size_t)DD * DD), 256, 0, stream>>>(Wb_p, bufW, DD, DD);
    gemm_wmma_bf16_k<<<ggrid, 256, 0, stream>>>(bufB, bufW, bb_p,
                                                bufS, nullptr, NN, DD, DD, 0, 0);
    // ReLU + BatchNorm (relu fused into stats + normalize)
    hipMemsetAsync(sums, 0, DD * sizeof(float), stream);
    hipMemsetAsync(sq,   0, DD * sizeof(float), stream);
    bn_stats_k<<<(NN + 511) / 512, 256, 0, stream>>>(bufS, sums, sq, NN);
    bn_norm_k<<<gblocks((size_t)NN * DD), 256, 0, stream>>>(bufS, sums, sq, g_p, be_p, bufH, NN);
  }

  // global mean pool + fc + tanh
  hipMemsetAsync(pool, 0, (size_t)NG * DD * sizeof(float), stream);
  hipMemsetAsync(cnt,  0, NG * sizeof(float), stream);
  pool_accum_k<<<gblocks((size_t)NN * DD), 256, 0, stream>>>(bufH, batch, pool, cnt);
  pool_final_k<<<gblocks((size_t)NG * DD), 256, 0, stream>>>(pool, cnt, p16);
  f32_to_bf16_T_k<<<gblocks((size_t)DD * DD), 256, 0, stream>>>(fcW, bufW, DD, DD);
  gemm_wmma_bf16_k<<<dim3((NG + BM - 1) / BM, DD / BN), 256, 0, stream>>>(
      p16, bufW, fcb, out, nullptr, NG, DD, DD, /*tanh*/2, 0);
}
